// GPT_55972013802085
// MI455X (gfx1250) — matrix operand order repaired and checked by reading, not compiled, not run on hardware
//
#include <hip/hip_runtime.h>
#include <hip/hip_bf16.h>

typedef __attribute__((ext_vector_type(16))) __bf16 v16bf;
typedef __attribute__((ext_vector_type(8)))  float  v8f;
typedef __attribute__((ext_vector_type(4)))  unsigned int v4u;
typedef __attribute__((ext_vector_type(8)))  int v8i;
typedef __attribute__((ext_vector_type(4)))  int v4i;

union ABFrag {
    v16bf v;
    uint4 q[2];
    __bf16 s[16];
};

__device__ inline v8f wmma_bf16(v16bf a, v16bf b, v8f c) {
    return __builtin_amdgcn_wmma_f32_16x16x32_bf16(false, a, false, b, (short)0, c, false, false);
}

// async copy 16B global -> LDS (CDNA5, ASYNCcnt-tracked, no VGPR roundtrip)
__device__ inline void async_copy_b128(const void* gptr, void* lptr) {
    unsigned lds = (unsigned)(unsigned long long)lptr;
    asm volatile("global_load_async_to_lds_b128 %0, %1, off"
                 :: "v"(lds), "v"(gptr) : "memory");
}
__device__ inline void wait_async_0() { asm volatile("s_wait_asynccnt 0x0" ::: "memory"); }
__device__ inline void wait_ds_0()    { asm volatile("s_wait_dscnt 0x0"    ::: "memory"); }

// ---------------------------------------------------------------------------
// Tensor Data Mover: 2D tile load Global -> LDS (bf16 elements).
// D# built per CDNA5 ISA ch.8: group0 = {count, lds_addr, global_addr, type=2},
// group1 = {data_size=2B, tensor dims, tile dims, dim0 stride}; groups 2/3
// zero (<=2D tensor). Completion tracked with TENSORcnt (per wave, in order).
// ---------------------------------------------------------------------------
__device__ inline void tdm_load_2d(const void* gptr, void* lptr,
                                   unsigned tile_d0,   // elements per row
                                   unsigned tile_d1,   // rows
                                   unsigned stride_d0) // row stride in elements
{
    unsigned long long ga = (unsigned long long)gptr;
    unsigned lds = (unsigned)(unsigned long long)lptr;
    unsigned tensor_d0 = stride_d0;      // full-width tiles: never OOB in dim0
    unsigned tensor_d1 = 0x100000u;      // large: never OOB in dim1
    v4u g0;
    g0[0] = 1u;                                            // count=1 (user D#)
    g0[1] = lds;                                           // lds_addr [63:32]
    g0[2] = (unsigned)(ga & 0xffffffffu);                  // global_addr[31:0]
    g0[3] = (unsigned)((ga >> 32) & 0x01ffffffu)           // global_addr[56:32]
          | (2u << 30);                                    // type=2 ("image")
    v8i g1;
    g1[0] = (int)(1u << 16);                               // data_size=1 -> 2B
    g1[1] = (int)((tensor_d0 & 0xffffu) << 16);            // tensor_dim0[15:0]
    g1[2] = (int)(((tensor_d0 >> 16) & 0xffffu)            // tensor_dim0[31:16]
          |        ((tensor_d1 & 0xffffu) << 16));         // tensor_dim1[15:0]
    g1[3] = (int)(((tensor_d1 >> 16) & 0xffffu)            // tensor_dim1[31:16]
          |        (tile_d0 << 16));                       // tile_dim0
    g1[4] = (int)tile_d1;                                  // tile_dim1 (tile_dim2=0)
    g1[5] = (int)stride_d0;                                // tensor_dim0_stride[31:0]
    g1[6] = 0;                                             // stride hi / dim1_stride
    g1[7] = 0;
    v4i z4 = {0, 0, 0, 0};
    v8i z8 = {0, 0, 0, 0, 0, 0, 0, 0};
    __builtin_amdgcn_tensor_load_to_lds(g0, g1, z4, z4, z8, 0);
}

// ---------------------------------------------------------------------------
// fp32 -> bf16 conversion (with optional zero padding past n up to npad)
// ---------------------------------------------------------------------------
__global__ __launch_bounds__(256) void cvt_kernel(const float* __restrict__ src,
                                                  __bf16* __restrict__ dst,
                                                  long n, long npad) {
    long i = (long)blockIdx.x * blockDim.x + threadIdx.x;
    long stride = (long)gridDim.x * blockDim.x;
    for (; i < npad; i += stride)
        dst[i] = (i < n) ? (__bf16)src[i] : (__bf16)0.0f;
}

// ---------------------------------------------------------------------------
// fp32 [K,N] -> bf16 [N,K] transposed conversion (per-layer via blockIdx.z)
// ---------------------------------------------------------------------------
__global__ __launch_bounds__(256) void cvtT_kernel(const float* __restrict__ src,
                                                   __bf16* __restrict__ dst,
                                                   int K, int N) {
    __shared__ __bf16 tile[32][33];
    src += (long)blockIdx.z * K * N;
    dst += (long)blockIdx.z * N * K;
    int n0 = blockIdx.x * 32, k0 = blockIdx.y * 32;
    int tx = threadIdx.x, ty = threadIdx.y;          // (32, 8)
#pragma unroll
    for (int i = ty; i < 32; i += 8)
        tile[i][tx] = (__bf16)src[(long)(k0 + i) * N + n0 + tx];
    __syncthreads();
#pragma unroll
    for (int i = ty; i < 32; i += 8)
        dst[(long)(n0 + i) * K + k0 + tx] = tile[tx][i];
}

// ---------------------------------------------------------------------------
// x[m,c] = emb[idx[m],c] + pos_emb[m%1024,c]
// ---------------------------------------------------------------------------
__global__ __launch_bounds__(256) void embed_kernel(const int* __restrict__ idx,
                                                    const float* __restrict__ emb,
                                                    const float* __restrict__ pos,
                                                    float* __restrict__ x) {
    int m = blockIdx.x;          // 0..2047
    int t = m & 1023;
    int tok = idx[m];
    const float* e = emb + (long)tok * 768;
    const float* p = pos + (long)t * 768;
    float* o = x + (long)m * 768;
    for (int c = threadIdx.x; c < 768; c += 256) o[c] = e[c] + p[c];
}

// ---------------------------------------------------------------------------
// LayerNorm over C=768, output bf16 (A operand for the following GEMM)
// ---------------------------------------------------------------------------
__global__ __launch_bounds__(256) void layernorm_kernel(const float* __restrict__ x,
                                                        const float* __restrict__ w,
                                                        const float* __restrict__ b,
                                                        __bf16* __restrict__ out) {
    __shared__ float s1[256];
    __shared__ float s2[256];
    int m = blockIdx.x;
    const float* row = x + (long)m * 768;
    float sum = 0.f, sq = 0.f;
    for (int c = threadIdx.x; c < 768; c += 256) { float v = row[c]; sum += v; sq += v * v; }
    s1[threadIdx.x] = sum; s2[threadIdx.x] = sq;
    __syncthreads();
    for (int s = 128; s > 0; s >>= 1) {
        if (threadIdx.x < s) { s1[threadIdx.x] += s1[threadIdx.x + s]; s2[threadIdx.x] += s2[threadIdx.x + s]; }
        __syncthreads();
    }
    float mu   = s1[0] * (1.f / 768.f);
    float var  = s2[0] * (1.f / 768.f) - mu * mu;
    float rstd = rsqrtf(var + 1e-5f);
    __bf16* o = out + (long)m * 768;
    for (int c = threadIdx.x; c < 768; c += 256)
        o[c] = (__bf16)((row[c] - mu) * rstd * w[c] + b[c]);
}

// ---------------------------------------------------------------------------
// bf16 GEMM: out[m,n] = act(A[m,:] @ B[n,:]^T + bias[n]) + resid[m,n]
//   A: [M,K] row-major; B: [N,K] row-major (weights pre-transposed once).
//   128x64 output tile per block (128 threads = 4 waves; each wave: 32x64).
//   Double-buffered LDS staged by the Tensor Data Mover: each wave issues
//   2 TDM tile loads per k-tile (its A quarter + B quarter) and waits on its
//   own TENSORcnt before the block barrier.
//   act: 0 = none, 1 = exact GELU
// ---------------------------------------------------------------------------
__global__ __launch_bounds__(128) void gemm_kernel(const __bf16* __restrict__ A, int lda,
                                                   const __bf16* __restrict__ B, int ldb,
                                                   int K,
                                                   const float* __restrict__ bias,
                                                   const float* __restrict__ resid,
                                                   float* __restrict__ outF,
                                                   __bf16* __restrict__ outB,
                                                   int ldo, int nstore, int act) {
    __shared__ __bf16 sA[2][128][32];   // [m][k]
    __shared__ __bf16 sBt[2][64][32];   // [n][k]
    int tid  = threadIdx.x;
    int lane = tid & 31;
    int wv   = tid >> 5;             // 0..3: wave id (uniform per wave)
    int hl   = lane >> 4;            // half-wave
    int mrow = lane & 15;
    int n0 = blockIdx.x * 64;
    int m0 = blockIdx.y * 128;

    v8f vzero = {0.f, 0.f, 0.f, 0.f, 0.f, 0.f, 0.f, 0.f};
    v8f acc0[4], acc1[4];
#pragma unroll
    for (int j = 0; j < 4; ++j) { acc0[j] = vzero; acc1[j] = vzero; }

    // each wave DMA-loads its 32-row A quarter and 16-row B quarter (2 TDMs)
    auto issue = [&](int buf, int k0) {
        tdm_load_2d(A + (long)(m0 + 32 * wv) * lda + k0, &sA[buf][32 * wv][0],
                    /*tile_d0=*/32, /*tile_d1=*/32, /*stride=*/(unsigned)lda);
        tdm_load_2d(B + (long)(n0 + 16 * wv) * ldb + k0, &sBt[buf][16 * wv][0],
                    /*tile_d0=*/32, /*tile_d1=*/16, /*stride=*/(unsigned)ldb);
    };

    issue(0, 0);
    int buf = 0;
    for (int k0 = 0; k0 < K; k0 += 32, buf ^= 1) {
        bool more = (k0 + 32 < K);
        if (more) {
            issue(buf ^ 1, k0 + 32);              // overlap next tile's DMA
            __builtin_amdgcn_s_wait_tensorcnt(2); // older pair (this buf) done
        } else {
            __builtin_amdgcn_s_wait_tensorcnt(0);
        }
        __syncthreads();

        // A fragments: rows 32*wv + {mrow, 16+mrow}; K(e) = e<8 ? 8*hl+e : 16+8*hl+(e-8)
        ABFrag a0, a1;
        a0.q[0] = *(const uint4*)(&sA[buf][32 * wv + mrow][8 * hl]);
        a0.q[1] = *(const uint4*)(&sA[buf][32 * wv + mrow][16 + 8 * hl]);
        a1.q[0] = *(const uint4*)(&sA[buf][32 * wv + 16 + mrow][8 * hl]);
        a1.q[1] = *(const uint4*)(&sA[buf][32 * wv + 16 + mrow][16 + 8 * hl]);
#pragma unroll
        for (int j = 0; j < 4; ++j) {
            // B fragment: col N = mrow, K(e) = e<8 ? 8*hl+e : 16+8*hl+(e-8)
            ABFrag b;
            b.q[0] = *(const uint4*)(&sBt[buf][16 * j + mrow][8 * hl]);
            b.q[1] = *(const uint4*)(&sBt[buf][16 * j + mrow][16 + 8 * hl]);
            acc0[j] = wmma_bf16(a0.v, b.v, acc0[j]);
            acc1[j] = wmma_bf16(a1.v, b.v, acc1[j]);
        }
        __syncthreads();   // all waves done reading buf before it is re-filled
    }

    // epilogue: D layout lane N = mrow, rows M = 8*hl + r
#pragma unroll
    for (int strip = 0; strip < 2; ++strip) {
#pragma unroll
        for (int j = 0; j < 4; ++j) {
            int col = n0 + 16 * j + mrow;
            if (col >= nstore) continue;
            float bv = bias ? bias[col] : 0.f;
#pragma unroll
            for (int r = 0; r < 8; ++r) {
                int m = m0 + 32 * wv + 16 * strip + 8 * hl + r;
                float v = (strip ? acc1[j][r] : acc0[j][r]) + bv;
                if (act == 1) v = 0.5f * v * (1.f + erff(v * 0.70710678118654752f));
                if (resid) v += resid[(long)m * ldo + col];
                if (outF) outF[(long)m * ldo + col] = v;
                else      outB[(long)m * ldo + col] = (__bf16)v;
            }
        }
    }
}

// ---------------------------------------------------------------------------
// Flash attention: 1 wave per (b, h, 16-query tile); key blocks of 32.
// qkv: [2048, 2304] bf16 (q | k | v each 768 wide); out: [2048, 768] bf16
// K tiles staged via global_load_async_to_lds_b128 (overlapped with the
// in-register V transpose); P row-sums via a WMMA against an all-ones frag.
// ---------------------------------------------------------------------------
__global__ __launch_bounds__(32) void attn_kernel(const __bf16* __restrict__ qkv,
                                                  __bf16* __restrict__ out) {
    __shared__ __bf16 sK[32][64];    // [key][c]
    __shared__ __bf16 sVt[64][32];   // [c][key]
    __shared__ __bf16 sP[16][32];    // [q][key]
    const int LDQ = 2304;
    int lane = threadIdx.x;
    int hl   = lane >> 4;
    int mrow = lane & 15;
    int qt = blockIdx.x;             // 0..63
    int bh = blockIdx.y;             // 0..23
    int bb = bh / 12, h = bh % 12;
    int qi0 = qt * 16;
    long rowbase = (long)bb * 1024;

    // preload Q fragments (A layout): e 0..7 -> k = p*32+8*hl+e, e 8..15 -> +16
    ABFrag aQ[2];
#pragma unroll
    for (int p = 0; p < 2; ++p) {
        const __bf16* qp = qkv + (rowbase + qi0 + mrow) * LDQ + h * 64 + p * 32;
        aQ[p].q[0] = *(const uint4*)(qp + 8 * hl);
        aQ[p].q[1] = *(const uint4*)(qp + 16 + 8 * hl);
    }
    // all-ones B fragment for row-sum WMMA
    ABFrag ones;
#pragma unroll
    for (int e = 0; e < 16; ++e) ones.s[e] = (__bf16)1.0f;

    float mrun[8], lrun[8], alpha[8];
#pragma unroll
    for (int r = 0; r < 8; ++r) { mrun[r] = -1e30f; lrun[r] = 0.f; }
    v8f vzero = {0.f, 0.f, 0.f, 0.f, 0.f, 0.f, 0.f, 0.f};
    v8f oacc[4];
#pragma unroll
    for (int j = 0; j < 4; ++j) oacc[j] = vzero;

    int kbmax = (qi0 + 15) >> 5;
    for (int kb = 0; kb <= kbmax; ++kb) {
        int kk0 = kb * 32;
        long base = (rowbase + kk0 + lane) * (long)LDQ + h * 64;
        // prior LDS reads must be complete before async writes can land in sK
        wait_ds_0();
        // async stage K tile [32][64]: lane owns key row = lane (8 x b128)
        {
            const __bf16* kp = qkv + base + 768;
#pragma unroll
            for (int q = 0; q < 8; ++q)
                async_copy_b128(kp + 8 * q, &sK[lane][8 * q]);
        }
        // manual transpose-stage V tile [64][32] while K copies are in flight
        {
            const __bf16* vp = qkv + base + 1536;
#pragma unroll
            for (int c = 0; c < 64; c += 8) {
                uint4 d = *(const uint4*)(vp + c);
                const __bf16* e = (const __bf16*)&d;
#pragma unroll
                for (int u = 0; u < 8; ++u) sVt[c + u][lane] = e[u];
            }
        }
        wait_async_0();
        __syncthreads();

        // S = Q @ K^T  (two 16-key groups g, head dim chained over p)
        v8f sacc[2] = {vzero, vzero};
#pragma unroll
        for (int g = 0; g < 2; ++g)
#pragma unroll
            for (int p = 0; p < 2; ++p) {
                ABFrag bK;   // B[c, key]: lane key = mrow (+16g), K(e) = p*32 + (e<8 ? 8*hl+e : 16+8*hl+(e-8))
                bK.q[0] = *(const uint4*)(&sK[16 * g + mrow][p * 32 + 8 * hl]);
                bK.q[1] = *(const uint4*)(&sK[16 * g + mrow][p * 32 + 16 + 8 * hl]);
                sacc[g] = wmma_bf16(aQ[p].v, bK.v, sacc[g]);
            }

        // scale + causal mask + online max (rows M = 8*hl + r)
#pragma unroll
        for (int r = 0; r < 8; ++r) {
            float s0 = sacc[0][r] * 0.125f;    // 1/sqrt(64)
            float s1 = sacc[1][r] * 0.125f;
            int qidx = qi0 + 8 * hl + r;
            if (kk0 + mrow      > qidx) s0 = -1e30f;
            if (kk0 + 16 + mrow > qidx) s1 = -1e30f;
            float mx = fmaxf(s0, s1);
#pragma unroll
            for (int d = 8; d >= 1; d >>= 1) mx = fmaxf(mx, __shfl_xor(mx, d, 32));
            float mnew = fmaxf(mrun[r], mx);
            alpha[r] = __expf(mrun[r] - mnew);
            mrun[r] = mnew;
            sP[8 * hl + r][mrow]      = (__bf16)__expf(s0 - mnew);
            sP[8 * hl + r][16 + mrow] = (__bf16)__expf(s1 - mnew);
        }
#pragma unroll
        for (int j = 0; j < 4; ++j)
#pragma unroll
            for (int r = 0; r < 8; ++r) oacc[j][r] *= alpha[r];
        __syncthreads();

        // O += P @ V (K = 32 keys); row-sums via P @ ones
        ABFrag aP;
        aP.q[0] = *(const uint4*)(&sP[mrow][8 * hl]);
        aP.q[1] = *(const uint4*)(&sP[mrow][16 + 8 * hl]);
        v8f rowsum = wmma_bf16(aP.v, ones.v, vzero);
#pragma unroll
        for (int j = 0; j < 4; ++j) {
            ABFrag bV;   // B[key, c]: lane c = mrow (+16j), K(e) = e<8 ? 8*hl+e : 16+8*hl+(e-8)
            bV.q[0] = *(const uint4*)(&sVt[16 * j + mrow][8 * hl]);
            bV.q[1] = *(const uint4*)(&sVt[16 * j + mrow][16 + 8 * hl]);
            oacc[j] = wmma_bf16(aP.v, bV.v, oacc[j]);
        }
#pragma unroll
        for (int r = 0; r < 8; ++r) lrun[r] = lrun[r] * alpha[r] + rowsum[r];
        __syncthreads();
    }

    // normalize and write bf16 [2048, 768]
#pragma unroll
    for (int r = 0; r < 8; ++r) {
        float inv = 1.f / lrun[r];
        long orow = (rowbase + qi0 + 8 * hl + r) * 768 + h * 64;
#pragma unroll
        for (int j = 0; j < 4; ++j)
            out[orow + 16 * j + mrow] = (__bf16)(oacc[j][r] * inv);
    }
}

// ---------------------------------------------------------------------------
// host orchestration
// ---------------------------------------------------------------------------
extern "C" void kernel_launch(void* const* d_in, const int* in_sizes, int n_in,
                              void* d_out, int out_size, void* d_ws, size_t ws_size,
                              hipStream_t stream) {
    const int*   idx    = (const int*)  d_in[0];
    const float* emb    = (const float*)d_in[1];
    const float* pos    = (const float*)d_in[2];
    const float* Wqkv   = (const float*)d_in[3];
    const float* Wproj  = (const float*)d_in[4];
    const float* ln1_w  = (const float*)d_in[5];
    const float* ln1_b  = (const float*)d_in[6];
    const float* ln2_w  = (const float*)d_in[7];
    const float* ln2_b  = (const float*)d_in[8];
    const float* W1     = (const float*)d_in[9];
    const float* b1     = (const float*)d_in[10];
    const float* W2     = (const float*)d_in[11];
    const float* b2     = (const float*)d_in[12];
    const float* lnf_w  = (const float*)d_in[13];
    const float* lnf_b  = (const float*)d_in[14];

    const int L = 6, C = 768, V = 50257, V_PAD = 50304; // V_PAD multiple of 64

    char* ws = (char*)d_ws;
    size_t off = 0;
    auto take = [&](size_t bytes) -> void* {
        void* p = ws + off;
        off += (bytes + 255) & ~(size_t)255;
        return p;
    };
    __bf16* emb_bf   = (__bf16*)take((size_t)V_PAD * C * 2);     // [V_PAD, C]
    __bf16* wqkvT    = (__bf16*)take((size_t)L * 3 * C * C * 2); // [3C, C] per layer
    __bf16* wprojT   = (__bf16*)take((size_t)L * C * C * 2);     // [C, C]
    __bf16* w1T      = (__bf16*)take((size_t)L * 4 * C * C * 2); // [4C, C]
    __bf16* w2T      = (__bf16*)take((size_t)L * C * 4 * C * 2); // [C, 4C]
    float*  x        = (float*) take((size_t)2048 * C * 4);
    __bf16* ln_bf    = (__bf16*)take((size_t)2048 * C * 2);
    __bf16* qkv_bf   = (__bf16*)take((size_t)2048 * 3 * C * 2);
    __bf16* attn_bf  = (__bf16*)take((size_t)2048 * C * 2);
    __bf16* h_bf     = (__bf16*)take((size_t)2048 * 4 * C * 2);

    // embedding: plain bf16 conversion (already [N,K]), zero-padded to V_PAD
    cvt_kernel<<<2048, 256, 0, stream>>>(emb, emb_bf, (long)V * C, (long)V_PAD * C);
    // weights: fused transpose + bf16 conversion -> [N,K]
    cvtT_kernel<<<dim3(3 * C / 32, C / 32, L),     dim3(32, 8), 0, stream>>>(Wqkv,  wqkvT,  C,     3 * C);
    cvtT_kernel<<<dim3(C / 32,     C / 32, L),     dim3(32, 8), 0, stream>>>(Wproj, wprojT, C,     C);
    cvtT_kernel<<<dim3(4 * C / 32, C / 32, L),     dim3(32, 8), 0, stream>>>(W1,    w1T,    C,     4 * C);
    cvtT_kernel<<<dim3(C / 32,     4 * C / 32, L), dim3(32, 8), 0, stream>>>(W2,    w2T,    4 * C, C);

    embed_kernel<<<2048, 256, 0, stream>>>(idx, emb, pos, x);

    for (int l = 0; l < L; ++l) {
        // qkv = LN1(x) @ Wqkv[l]   (bf16 out)
        layernorm_kernel<<<2048, 256, 0, stream>>>(x, ln1_w + l * C, ln1_b + l * C, ln_bf);
        gemm_kernel<<<dim3(36, 16), 128, 0, stream>>>(
            ln_bf, C, wqkvT + (size_t)l * 3 * C * C, C, C,
            nullptr, nullptr, nullptr, qkv_bf, 3 * C, 3 * C, 0);
        // attention
        attn_kernel<<<dim3(64, 24), 32, 0, stream>>>(qkv_bf, attn_bf);
        // x += attn @ Wproj[l]
        gemm_kernel<<<dim3(12, 16), 128, 0, stream>>>(
            attn_bf, C, wprojT + (size_t)l * C * C, C, C,
            nullptr, x, x, nullptr, C, C, 0);
        // h = gelu(LN2(x) @ W1[l] + b1[l])   (bf16 out)
        layernorm_kernel<<<2048, 256, 0, stream>>>(x, ln2_w + l * C, ln2_b + l * C, ln_bf);
        gemm_kernel<<<dim3(48, 16), 128, 0, stream>>>(
            ln_bf, C, w1T + (size_t)l * 4 * C * C, C, C,
            b1 + l * 4 * C, nullptr, nullptr, h_bf, 4 * C, 4 * C, 1);
        // x += h @ W2[l] + b2[l]
        gemm_kernel<<<dim3(12, 16), 128, 0, stream>>>(
            h_bf, 4 * C, w2T + (size_t)l * C * 4 * C, 4 * C, 4 * C,
            b2 + l * C, x, x, nullptr, C, C, 0);
    }

    // logits = LN_f(x) @ emb^T  (emb is already [V, C] = [N, K])
    layernorm_kernel<<<2048, 256, 0, stream>>>(x, lnf_w, lnf_b, ln_bf);
    gemm_kernel<<<dim3(V_PAD / 64, 16), 128, 0, stream>>>(
        ln_bf, C, emb_bf, C, C,
        nullptr, nullptr, (float*)d_out, nullptr, V, V, 0);
}
